// GATConvModel_52501680226999
// MI455X (gfx1250) — matrix operand order repaired and checked by reading, not compiled, not run on hardware
//
#include <hip/hip_runtime.h>
#include <hip/hip_bf16.h>
#include <stdint.h>

#define N_NODES 20000
#define N_EDGES 300000
#define E_TOT   (N_EDGES + N_NODES)
#define DIN     128
#define DH      256
#define DOUT    128
#define NGRAPH  256
#define NEG_SLOPE 0.2f

typedef __attribute__((ext_vector_type(16))) __bf16 v16bf;
typedef __attribute__((ext_vector_type(8)))  float  v8f;

// ---------------- helpers ----------------

__device__ __forceinline__ float lrelu(float x) {
    return x > 0.f ? x : NEG_SLOPE * x;
}

// order-preserving float->uint encoding for atomic max on floats
__device__ __forceinline__ unsigned enc_f32(float f) {
    int i = __float_as_int(f);
    return (i >= 0) ? ((unsigned)i | 0x80000000u) : ~(unsigned)i;
}
__device__ __forceinline__ float dec_f32(unsigned u) {
    int i = (u & 0x80000000u) ? (int)(u & 0x7FFFFFFFu) : (int)(~u);
    return __int_as_float(i);
}
#define ENC_NEG_INF 0x007FFFFFu  // enc(-inf)

__device__ __forceinline__ void edge_sd(const long long* __restrict__ ei, int e,
                                        int& si, int& di) {
    if (e < N_EDGES) { si = (int)ei[e]; di = (int)ei[N_EDGES + e]; }
    else             { si = di = e - N_EDGES; }   // self-loops appended
}

// ---------------- generic utility kernels ----------------

__global__ void zero_f32(float* __restrict__ p, int n) {
    int i = blockIdx.x * blockDim.x + threadIdx.x;
    if (i < n) p[i] = 0.f;
}

// W [K x N] f32 row-major -> Wt [N x K] bf16 row-major (K contiguous)
__global__ void conv_wt(const float* __restrict__ W, __bf16* __restrict__ Wt,
                        int K, int N) {
    int i = blockIdx.x * blockDim.x + threadIdx.x;
    if (i >= K * N) return;
    int k = i / N, n = i % N;
    Wt[(size_t)n * K + k] = (__bf16)W[i];
}

// ---------------- WMMA GEMM: C[M,N] = act(A[M,K] @ Bt[N,K]^T + bias) ----------------
// A: f32 row-major (converted bf16 on the fly, unconditional b128 loads).
// Bt: bf16, row n = column n of the weight, K contiguous.
// Block = 128 threads = 4 waves; wave w owns NT consecutive 16x16 n-tiles,
// reusing the A fragment NT times (NT back-to-back v_wmma per K-step).
// REQUIRES: M % 16 == 0, N == 64*NT, K % 32 == 0  (true for all call sites).

template <int NT>
__global__ void gemm_bf16(const float* __restrict__ A, const __bf16* __restrict__ Bt,
                          const float* __restrict__ bias, float* __restrict__ C,
                          int M, int N, int K, int act) {
    const int lane = threadIdx.x & 31;
    const int wave = threadIdx.x >> 5;
    const int m0   = blockIdx.x * 16;
    const int half = lane >> 4;      // 0: lanes 0-15, 1: lanes 16-31
    const int lr   = lane & 15;
    const int nb   = wave * NT * 16; // first column of this wave's tile group

    const float*  __restrict__ arow  = A  + (size_t)(m0 + lr) * K;
    const __bf16* __restrict__ brow0 = Bt + (size_t)(nb + lr) * K;

    v8f c[NT];
#pragma unroll
    for (int t = 0; t < NT; ++t) c[t] = (v8f){};

    for (int k0 = 0; k0 < K; k0 += 32) {
        // A 16x32 bf16 fragment: half=0 lanes hold K {0..15},
        // half=1 lanes hold K {16..31} (matches B); 2 values per VGPR.
        const int ka = k0 + half * 16;
        const float4* ap = (const float4*)(arow + ka);
        float4 f0 = ap[0], f1 = ap[1];          // K = ka .. ka+7
        const float4* ap2 = (const float4*)(arow + ka + 8);
        float4 f2 = ap2[0], f3 = ap2[1];        // K = ka+8 .. ka+15
        v16bf a;
        a[0]  = (__bf16)f0.x; a[1]  = (__bf16)f0.y; a[2]  = (__bf16)f0.z; a[3]  = (__bf16)f0.w;
        a[4]  = (__bf16)f1.x; a[5]  = (__bf16)f1.y; a[6]  = (__bf16)f1.z; a[7]  = (__bf16)f1.w;
        a[8]  = (__bf16)f2.x; a[9]  = (__bf16)f2.y; a[10] = (__bf16)f2.z; a[11] = (__bf16)f2.w;
        a[12] = (__bf16)f3.x; a[13] = (__bf16)f3.y; a[14] = (__bf16)f3.z; a[15] = (__bf16)f3.w;

        // B 32x16 fragment per tile: lane = column, half selects K block of 16.
        const int kb = k0 + half * 16;
#pragma unroll
        for (int t = 0; t < NT; ++t) {
            v16bf b = *(const v16bf*)(brow0 + (size_t)t * 16 * K + kb);
            c[t] = __builtin_amdgcn_wmma_f32_16x16x32_bf16(
                       false, a, false, b, (short)0, c[t], false, false);
        }
    }

    // C/D layout: lane -> n = ntile*16 + (lane&15); VGPR v -> m = m0 + v + 8*half
#pragma unroll
    for (int t = 0; t < NT; ++t) {
        const int nOut = nb + t * 16 + lr;
        const float bv = bias ? bias[nOut] : 0.f;
        const int mBase = m0 + 8 * half;
#pragma unroll
        for (int v = 0; v < 8; ++v) {
            float o = c[t][v] + bv;
            if (act) o = fmaxf(o, 0.f);
            C[(size_t)(mBase + v) * N + nOut] = o;
        }
    }
}

// ---------------- GAT attention kernels ----------------

// per-node attention scores: s[n] = h[n].a_src, d[n] = h[n].a_dst  (wave per node)
__global__ void node_scores(const float* __restrict__ h,
                            const float* __restrict__ a_s, const float* __restrict__ a_d,
                            float* __restrict__ s, float* __restrict__ d) {
    int gw = (blockIdx.x * blockDim.x + threadIdx.x) >> 5;
    int lane = threadIdx.x & 31;
    if (gw >= N_NODES) return;
    const float* hr = h + (size_t)gw * DH;
    float ss = 0.f, dd = 0.f;
#pragma unroll
    for (int j = 0; j < 8; ++j) {
        int f = lane + 32 * j;
        float v = hr[f];
        ss += v * a_s[f];
        dd += v * a_d[f];
    }
#pragma unroll
    for (int off = 16; off > 0; off >>= 1) {
        ss += __shfl_down(ss, off);
        dd += __shfl_down(dd, off);
    }
    if (lane == 0) { s[gw] = ss; d[gw] = dd; }
}

__global__ void init_softmax(unsigned* __restrict__ m_enc, float* __restrict__ denom) {
    int i = blockIdx.x * blockDim.x + threadIdx.x;
    if (i < N_NODES) { m_enc[i] = ENC_NEG_INF; denom[i] = 0.f; }
}

__global__ void edge_max(const long long* __restrict__ ei,
                         const float* __restrict__ s, const float* __restrict__ d,
                         unsigned* __restrict__ m_enc) {
    int e = blockIdx.x * blockDim.x + threadIdx.x;
    if (e >= E_TOT) return;
    int si, di; edge_sd(ei, e, si, di);
    float ev = lrelu(s[si] + d[di]);
    atomicMax(&m_enc[di], enc_f32(ev));
}

__global__ void decode_max(const unsigned* __restrict__ m_enc, float* __restrict__ m_f) {
    int i = blockIdx.x * blockDim.x + threadIdx.x;
    if (i < N_NODES) m_f[i] = dec_f32(m_enc[i]);
}

__global__ void edge_sumexp(const long long* __restrict__ ei,
                            const float* __restrict__ s, const float* __restrict__ d,
                            const float* __restrict__ m_f, float* __restrict__ denom) {
    int e = blockIdx.x * blockDim.x + threadIdx.x;
    if (e >= E_TOT) return;
    int si, di; edge_sd(ei, e, si, di);
    float ev = lrelu(s[si] + d[di]);
    atomicAdd(&denom[di], expf(ev - m_f[di]));
}

// wave per edge: out[dst] += alpha * h[src]   (8 feats per lane)
__global__ void edge_agg(const long long* __restrict__ ei,
                         const float* __restrict__ s, const float* __restrict__ d,
                         const float* __restrict__ m_f, const float* __restrict__ denom,
                         const float* __restrict__ h, float* __restrict__ outx) {
    int gw = (blockIdx.x * blockDim.x + threadIdx.x) >> 5;
    int lane = threadIdx.x & 31;
    if (gw >= E_TOT) return;
    int si, di; edge_sd(ei, gw, si, di);
    float ev = lrelu(s[si] + d[di]);
    float alpha = expf(ev - m_f[di]) / denom[di];
    const float* hs = h + (size_t)si * DH + lane * 8;
    float* od = outx + (size_t)di * DH + lane * 8;
#pragma unroll
    for (int j = 0; j < 8; ++j)
        atomicAdd(&od[j], alpha * hs[j]);
}

__global__ void bias_relu(float* __restrict__ x, const float* __restrict__ b, int n) {
    int i = blockIdx.x * blockDim.x + threadIdx.x;
    if (i < n) x[i] = fmaxf(x[i] + b[i & (DH - 1)], 0.f);
}

// ---------------- pooling ----------------

__global__ void pool_scatter(const float* __restrict__ h, const long long* __restrict__ batch,
                             float* __restrict__ g_sum, float* __restrict__ cnt) {
    int gw = (blockIdx.x * blockDim.x + threadIdx.x) >> 5;
    int lane = threadIdx.x & 31;
    if (gw >= N_NODES) return;
    int b = (int)batch[gw];
    const float* hr = h + (size_t)gw * DH + lane * 8;
    float* gr = g_sum + (size_t)b * DH + lane * 8;
#pragma unroll
    for (int j = 0; j < 8; ++j)
        atomicAdd(&gr[j], hr[j]);
    if (lane == 0) atomicAdd(&cnt[b], 1.0f);
}

__global__ void pool_norm(const float* __restrict__ g_sum, const float* __restrict__ cnt,
                          float* __restrict__ g) {
    int i = blockIdx.x * blockDim.x + threadIdx.x;
    if (i >= NGRAPH * DH) return;
    float c = cnt[i / DH];
    g[i] = g_sum[i] / fmaxf(c, 1.0f);
}

// ---------------- host side ----------------

static void run_gat_layer(const float* A, int K, const __bf16* Wt,
                          const float* a_s, const float* a_d, const float* b,
                          float* h, float* outx,
                          float* s, float* d, unsigned* m_enc, float* m_f, float* denom,
                          const long long* ei, hipStream_t stream) {
    // N = DH = 256 -> NT = 4
    gemm_bf16<4><<<N_NODES / 16, 128, 0, stream>>>(A, Wt, nullptr, h, N_NODES, DH, K, 0);
    node_scores<<<(N_NODES + 7) / 8, 256, 0, stream>>>(h, a_s, a_d, s, d);
    init_softmax<<<(N_NODES + 255) / 256, 256, 0, stream>>>(m_enc, denom);
    edge_max<<<(E_TOT + 255) / 256, 256, 0, stream>>>(ei, s, d, m_enc);
    decode_max<<<(N_NODES + 255) / 256, 256, 0, stream>>>(m_enc, m_f);
    edge_sumexp<<<(E_TOT + 255) / 256, 256, 0, stream>>>(ei, s, d, m_f, denom);
    zero_f32<<<(N_NODES * DH + 255) / 256, 256, 0, stream>>>(outx, N_NODES * DH);
    edge_agg<<<(E_TOT + 7) / 8, 256, 0, stream>>>(ei, s, d, m_f, denom, h, outx);
    bias_relu<<<(N_NODES * DH + 255) / 256, 256, 0, stream>>>(outx, b, N_NODES * DH);
}

extern "C" void kernel_launch(void* const* d_in, const int* in_sizes, int n_in,
                              void* d_out, int out_size, void* d_ws, size_t ws_size,
                              hipStream_t stream) {
    const float*     x     = (const float*)d_in[0];
    const long long* ei    = (const long long*)d_in[1];
    const long long* batch = (const long long*)d_in[2];
    const float *W1 = (const float*)d_in[3],  *as1 = (const float*)d_in[4],
                *ad1 = (const float*)d_in[5], *b1  = (const float*)d_in[6];
    const float *W2 = (const float*)d_in[7],  *as2 = (const float*)d_in[8],
                *ad2 = (const float*)d_in[9], *b2  = (const float*)d_in[10];
    const float *W3 = (const float*)d_in[11], *as3 = (const float*)d_in[12],
                *ad3 = (const float*)d_in[13], *b3 = (const float*)d_in[14];
    const float *fc1w = (const float*)d_in[15], *fc1b = (const float*)d_in[16];
    const float *fc2w = (const float*)d_in[17], *fc2b = (const float*)d_in[18];
    const float *fc3w = (const float*)d_in[19], *fc3b = (const float*)d_in[20];

    char* ws = (char*)d_ws;
    size_t off = 0;
    auto take = [&](size_t bytes) -> char* {
        char* p = ws + off;
        off = (off + bytes + 255) & ~(size_t)255;
        return p;
    };
    float*    h      = (float*)take((size_t)N_NODES * DH * 4);   // GEMM output
    float*    bx     = (float*)take((size_t)N_NODES * DH * 4);   // agg / next-layer input
    float*    s      = (float*)take((size_t)N_NODES * 4);
    float*    d      = (float*)take((size_t)N_NODES * 4);
    unsigned* m_enc  = (unsigned*)take((size_t)N_NODES * 4);
    float*    m_f    = (float*)take((size_t)N_NODES * 4);
    float*    denom  = (float*)take((size_t)N_NODES * 4);
    __bf16*   w1t    = (__bf16*)take((size_t)DH * DIN * 2);
    __bf16*   w2t    = (__bf16*)take((size_t)DH * DH * 2);
    __bf16*   w3t    = (__bf16*)take((size_t)DH * DH * 2);
    __bf16*   f1t    = (__bf16*)take((size_t)DH * DH * 2);
    __bf16*   f2t    = (__bf16*)take((size_t)DH * DH * 2);
    __bf16*   f3t    = (__bf16*)take((size_t)DOUT * DH * 2);
    float*    g_sum  = (float*)take((size_t)NGRAPH * DH * 4);
    float*    cnt    = (float*)take((size_t)NGRAPH * 4);
    float*    g0     = (float*)take((size_t)NGRAPH * DH * 4);
    float*    g1     = (float*)take((size_t)NGRAPH * DH * 4);
    float*    g2     = (float*)take((size_t)NGRAPH * DH * 4);
    (void)ws_size; (void)in_sizes; (void)n_in; (void)out_size;

    // one-shot weight transpose + bf16 convert (deterministic, cheap)
    conv_wt<<<(DIN * DH + 255) / 256, 256, 0, stream>>>(W1, w1t, DIN, DH);
    conv_wt<<<(DH * DH + 255) / 256, 256, 0, stream>>>(W2, w2t, DH, DH);
    conv_wt<<<(DH * DH + 255) / 256, 256, 0, stream>>>(W3, w3t, DH, DH);
    conv_wt<<<(DH * DH + 255) / 256, 256, 0, stream>>>(fc1w, f1t, DH, DH);
    conv_wt<<<(DH * DH + 255) / 256, 256, 0, stream>>>(fc2w, f2t, DH, DH);
    conv_wt<<<(DH * DOUT + 255) / 256, 256, 0, stream>>>(fc3w, f3t, DH, DOUT);

    // 3 GAT layers (result of each layer lands in bx)
    run_gat_layer(x,  DIN, w1t, as1, ad1, b1, h, bx, s, d, m_enc, m_f, denom, ei, stream);
    run_gat_layer(bx, DH,  w2t, as2, ad2, b2, h, bx, s, d, m_enc, m_f, denom, ei, stream);
    run_gat_layer(bx, DH,  w3t, as3, ad3, b3, h, bx, s, d, m_enc, m_f, denom, ei, stream);

    // global mean pool
    zero_f32<<<(NGRAPH * DH + 255) / 256, 256, 0, stream>>>(g_sum, NGRAPH * DH);
    zero_f32<<<1, 256, 0, stream>>>(cnt, NGRAPH);
    pool_scatter<<<(N_NODES + 7) / 8, 256, 0, stream>>>(bx, batch, g_sum, cnt);
    pool_norm<<<(NGRAPH * DH + 255) / 256, 256, 0, stream>>>(g_sum, cnt, g0);

    // MLP head (WMMA GEMMs, bias+ReLU fused)
    gemm_bf16<4><<<NGRAPH / 16, 128, 0, stream>>>(g0, f1t, fc1b, g1, NGRAPH, DH, DH, 1);
    gemm_bf16<4><<<NGRAPH / 16, 128, 0, stream>>>(g1, f2t, fc2b, g2, NGRAPH, DH, DH, 1);
    gemm_bf16<2><<<NGRAPH / 16, 128, 0, stream>>>(g2, f3t, fc3b, (float*)d_out,
                                                  NGRAPH, DOUT, DH, 0);
}
